// Encoder_4131758538920
// MI455X (gfx1250) — matrix operand order repaired and checked by reading, not compiled, not run on hardware
//
#include <hip/hip_runtime.h>
#include <hip/hip_bf16.h>
#include <stdint.h>

// Problem constants (from reference)
#define BB   32
#define TT   512
#define EMBD 512
#define HH   1024
#define G3   3072   // 3*H
#define NWG  16     // workgroups per direction (each owns 64 H-columns)
#define KT_U 32     // K-tiles over U (H/32)
#define KT_W 16     // K-tiles over W (EMB/32)
#define KT_A 48     // total K-tiles per gate in packed B

// LDS row pitches (dwords): stride % 64 == 4 (h) or 12 (x) -> conflict-free across 16 batches,
// stride % 4 == 0 -> uint4-aligned rows.
#define HPITCH 516  // 512 dwords of h (1024 bf16) + 4 pad
#define XPITCH 268  // 256 dwords of x (512 bf16) + 12 pad

typedef __attribute__((ext_vector_type(16))) __bf16 v16bf;
typedef __attribute__((ext_vector_type(8)))  float  v8f;

union ABu { v16bf v; uint32_t u[8]; uint16_t s[16]; };

__device__ __forceinline__ uint16_t f2bf(float f) {
  union { float f; uint32_t u; } cv; cv.f = f;
  uint32_t u = cv.u;
  u += 0x7fffu + ((u >> 16) & 1u);          // round-to-nearest-even
  return (uint16_t)(u >> 16);
}

// Async global->LDS 16-byte copy (CDNA5 ASYNCcnt-tracked path).
// lds_off = byte offset into LDS (low 32 bits of generic shared pointer).
__device__ __forceinline__ void async_ld_b128(uint32_t lds_off, const void* gptr) {
  asm volatile("global_load_async_to_lds_b128 %0, %1, off"
               :: "v"(lds_off), "v"((uint64_t)(uintptr_t)gptr)
               : "memory");
}
__device__ __forceinline__ void wait_async0() {
  asm volatile("s_wait_asynccnt 0" ::: "memory");
}

// ---------------- prep kernels ----------------

__global__ void k_zero_f(float* p, size_t n) {
  size_t i = (size_t)blockIdx.x * blockDim.x + threadIdx.x;
  size_t st = (size_t)gridDim.x * blockDim.x;
  for (; i < n; i += st) p[i] = 0.0f;
}

// Pack U (H x 3H) and W (EMB x 3H) f32 weights into fragment-linear bf16:
//   dst[((n16*3 + g)*KT_A + kt)*512 + lane*16 + e]
//     = B element row k = kt*32 + (e<8 ? 8*(lane/16)+e : 16+8*(lane/16)+e-8)
//       (U for kt<32, W rows for kt>=32), col = g*H + n16*16 + lane%16
// Each WMMA B-fragment is a contiguous 1 KB block; per-(gate,kt) blocks are
// reachable from one base pointer via immediate offsets.
__global__ void k_pack_b(const float* __restrict__ U, const float* __restrict__ W,
                         uint16_t* __restrict__ dst) {
  const size_t n = (size_t)64 * 3 * KT_A * 512;
  size_t i = (size_t)blockIdx.x * blockDim.x + threadIdx.x;
  size_t st = (size_t)gridDim.x * blockDim.x;
  for (; i < n; i += st) {
    const int e    = (int)(i & 15);
    const int lane = (int)((i >> 4) & 31);
    const size_t j = i >> 9;               // fragment index = (n16*3+g)*KT_A + kt
    const int kt   = (int)(j % KT_A);
    const size_t r3 = j / KT_A;
    const int g    = (int)(r3 % 3);
    const int n16  = (int)(r3 / 3);
    const int col  = g * HH + n16 * 16 + (lane & 15);
    const int kk   = (e < 8) ? (8 * (lane >> 4) + e) : (16 + 8 * (lane >> 4) + e - 8);
    const float v = (kt < KT_U)
        ? U[(size_t)(kt * 32 + kk) * G3 + col]
        : W[(size_t)((kt - KT_U) * 32 + kk) * G3 + col];
    dst[i] = f2bf(v);
  }
}

// x_bf[b*T + t][e] = bf16(emb[tokens[b*T+t]][e])
__global__ void k_gather_x(const int* __restrict__ tok, const float* __restrict__ emb,
                           uint16_t* __restrict__ xb) {
  size_t i = (size_t)blockIdx.x * blockDim.x + threadIdx.x;
  size_t st = (size_t)gridDim.x * blockDim.x;
  const size_t n = (size_t)BB * TT * EMBD;
  for (; i < n; i += st) {
    size_t bt = i >> 9;            // EMBD == 512
    int    e  = (int)(i & 511);
    int    tk = tok[bt];
    xb[i] = f2bf(emb[(size_t)tk * EMBD + e]);
  }
}

// ---------------- fused bidirectional GRU ----------------
// grid = (NWG, 2 dirs), block = 256 (8 waves).
// Wave (mtile = wave/4, nsub = wave%4) computes the 16x16 output tile
// [batch mtile*16.., H-cols wg*64 + nsub*16..] for all three gates.
// Pipelined step:
//   issue async h->LDS | x-part GEMM (x staged last step) | wait+sync |
//   h-part GEMM | issue async x(t+1) | epilogue | wait | device barrier
__global__ __launch_bounds__(256)
void k_bigru(const uint16_t* __restrict__ xb,
             const uint16_t* __restrict__ Bpf, const uint16_t* __restrict__ Bpb,
             const float* __restrict__ bfin, const float* __restrict__ bfrc,
             const float* __restrict__ bbin, const float* __restrict__ bbrc,
             const unsigned char* __restrict__ mask,  // bool8 (B,T)
             float* __restrict__ hf,      // [2 dir][2 buf][B*H] f32 carry
             uint16_t* __restrict__ hbf,  // same, bf16 mirror (WMMA A source)
             float* __restrict__ out,     // [B][T][H], pre-zeroed; both dirs atomicAdd
             unsigned int* __restrict__ ctr) {  // [2] step barrier counters
  __shared__ uint32_t shh[BB * HPITCH];        // h_t staged (bf16), ~66 KB
  __shared__ uint32_t shx[2][BB * XPITCH];     // x double buffer, 2 x ~34 KB

  const int tid  = threadIdx.x;
  const int dir  = blockIdx.y;
  const int wg   = blockIdx.x;
  const int lane = tid & 31;
  const int wave = tid >> 5;
  const int mtile = wave >> 2;
  const int nsub  = wave & 3;
  const int half  = lane >> 4;
  const int l15   = lane & 15;

  const uint16_t* Bp = dir ? Bpb : Bpf;
  const float* b_in  = dir ? bbin : bfin;
  const float* b_rec = dir ? bbrc : bfrc;
  unsigned int* myctr = ctr + dir;
  float*    hbase  = hf  + (size_t)dir * 2 * BB * HH;
  uint16_t* hbbase = hbf + (size_t)dir * 2 * BB * HH;

  const int n16 = wg * 4 + nsub;         // this wave's 16-column tile index
  const int nl  = n16 * 16 + l15;        // this lane's H column (C/D: lane%16 = N)
  const float bz  = b_in[nl]          + b_rec[nl];
  const float br  = b_in[HH + nl]     + b_rec[HH + nl];
  const float bhi = b_in[2 * HH + nl];
  const float bhr = b_rec[2 * HH + nl];
  const int batchA = mtile * 16 + l15;         // A fragment: lane%16 = M row
  const int batchC = mtile * 16 + half * 8;    // C fragment: M = v + 8*(lane/16)

  // Per-wave packed-B base: fragments at immediate offsets (gate*KT_A + kt)*512.
  const uint16_t* Bwave = Bp + ((size_t)n16 * 3 * KT_A) * 512 + (size_t)lane * 16;

  // ---- prologue: stage x(time of t=0) into shx[0] ----
  {
    const int time0 = dir ? (TT - 1) : 0;
    for (int i = tid; i < BB * 64; i += 256) {
      const int b = i >> 6, c = i & 63;
      async_ld_b128((uint32_t)(uintptr_t)(&shx[0][b * XPITCH + c * 4]),
                    xb + ((size_t)b * TT + time0) * EMBD + c * 8);
    }
    wait_async0();
    __syncthreads();
  }

  for (int t = 0; t < TT; ++t) {
    const int time = dir ? (TT - 1 - t) : t;

    const float*    hr  = hbase  + (size_t)(t & 1) * BB * HH;
    const uint16_t* hrb = hbbase + (size_t)(t & 1) * BB * HH;
    float*    hw  = hbase  + (size_t)((t + 1) & 1) * BB * HH;
    uint16_t* hwb = hbbase + (size_t)((t + 1) & 1) * BB * HH;

    // (1) issue async stage of h_t (B x H bf16); completion hidden by x-GEMM
    for (int i = tid; i < BB * 128; i += 256) {
      const int b = i >> 7, c = i & 127;
      async_ld_b128((uint32_t)(uintptr_t)(shh + b * HPITCH + c * 4),
                    hrb + (size_t)b * HH + c * 8);
    }

    // Opaque zero offset: blocks LICM from hoisting (and then spilling) the
    // loop-invariant weight fragments across the 512-step loop.
    int licm = 0;
    asm volatile("" : "+v"(licm));
    const uint16_t* Bt = Bwave + licm;

    v8f az = {}, ar = {}, ahu = {}, ahx = {};

    // (2) input part: K = EMB over W (A = x_t staged during previous step)
    {
      const uint32_t* xrow = &shx[t & 1][batchA * XPITCH];
      #pragma unroll 2
      for (int kt = 0; kt < KT_W; ++kt) {
        ABu a;
        const int kb2 = kt * 16 + half * 4;  // dword index = K/2 (ISA A layout)
        #pragma unroll
        for (int i = 0; i < 4; ++i) {
          a.u[i]     = xrow[kb2 + i];
          a.u[i + 4] = xrow[kb2 + 8 + i];
        }
        ABu bz8, br8, bh8;
        bz8.v = *(const v16bf*)(Bt + (size_t)(KT_U + kt) * 512);
        br8.v = *(const v16bf*)(Bt + (size_t)(KT_A + KT_U + kt) * 512);
        bh8.v = *(const v16bf*)(Bt + (size_t)(2 * KT_A + KT_U + kt) * 512);
        az  = __builtin_amdgcn_wmma_f32_16x16x32_bf16(false, a.v, false, bz8.v, (short)0, az,  false, false);
        ar  = __builtin_amdgcn_wmma_f32_16x16x32_bf16(false, a.v, false, br8.v, (short)0, ar,  false, false);
        ahx = __builtin_amdgcn_wmma_f32_16x16x32_bf16(false, a.v, false, bh8.v, (short)0, ahx, false, false);
      }
    }

    // (3) h arrived: wait own async ops, then block-sync for LDS visibility
    wait_async0();
    __syncthreads();

    // (4) recurrent part: K = H over U (A = h_bf16 from LDS)
    {
      const uint32_t* hrow = shh + batchA * HPITCH;
      #pragma unroll 2
      for (int kt = 0; kt < KT_U; ++kt) {
        ABu a;
        const int kb2 = kt * 16 + half * 4;
        #pragma unroll
        for (int i = 0; i < 4; ++i) {
          a.u[i]     = hrow[kb2 + i];
          a.u[i + 4] = hrow[kb2 + 8 + i];
        }
        ABu bz8, br8, bh8;
        bz8.v = *(const v16bf*)(Bt + (size_t)kt * 512);
        br8.v = *(const v16bf*)(Bt + (size_t)(KT_A + kt) * 512);
        bh8.v = *(const v16bf*)(Bt + (size_t)(2 * KT_A + kt) * 512);
        az  = __builtin_amdgcn_wmma_f32_16x16x32_bf16(false, a.v, false, bz8.v, (short)0, az,  false, false);
        ar  = __builtin_amdgcn_wmma_f32_16x16x32_bf16(false, a.v, false, br8.v, (short)0, ar,  false, false);
        ahu = __builtin_amdgcn_wmma_f32_16x16x32_bf16(false, a.v, false, bh8.v, (short)0, ahu, false, false);
      }
    }

    // (5) prefetch next step's x into the other shx buffer (hidden by epilogue)
    if (t + 1 < TT) {
      const int tn = dir ? (TT - 2 - t) : (t + 1);
      uint32_t* shxn = &shx[(t + 1) & 1][0];
      for (int i = tid; i < BB * 64; i += 256) {
        const int b = i >> 6, c = i & 63;
        async_ld_b128((uint32_t)(uintptr_t)(shxn + b * XPITCH + c * 4),
                      xb + ((size_t)b * TT + tn) * EMBD + c * 8);
      }
    }

    // (6) gate nonlinearities + masked update + output accumulation
    #pragma unroll
    for (int v = 0; v < 8; ++v) {
      const int b = batchC + v;
      const float hold = hr[(size_t)b * HH + nl];
      const float z  = __builtin_amdgcn_rcpf(1.0f + __expf(-(az[v] + bz)));
      const float r  = __builtin_amdgcn_rcpf(1.0f + __expf(-(ar[v] + br)));
      const float y  = ahx[v] + bhi + r * (ahu[v] + bhr);
      // tanh(y) = 1 - 2/(1 + exp(2y))
      const float hh = 1.0f - 2.0f * __builtin_amdgcn_rcpf(1.0f + __expf(2.0f * y));
      const float hc = z * hold + (1.0f - z) * hh;
      const float hn = mask[(size_t)b * TT + time] ? hc : hold;
      hw [(size_t)b * HH + nl] = hn;
      hwb[(size_t)b * HH + nl] = f2bf(hn);
      atomicAdd(out + ((size_t)b * TT + time) * HH + nl, hn);  // fwd + bwd sum
    }

    // (7) close pipeline + device-scope per-direction step barrier
    wait_async0();          // next-x staging complete before anyone re-reads it
    __threadfence();
    __syncthreads();
    if (tid == 0) {
      __hip_atomic_fetch_add(myctr, 1u, __ATOMIC_RELEASE, __HIP_MEMORY_SCOPE_AGENT);
      const unsigned int target = (unsigned int)NWG * (unsigned int)(t + 1);
      while (__hip_atomic_load(myctr, __ATOMIC_ACQUIRE, __HIP_MEMORY_SCOPE_AGENT) < target) {
        __builtin_amdgcn_s_sleep(2);
      }
    }
    __syncthreads();
  }
}

// ---------------- host-side orchestration ----------------

extern "C" void kernel_launch(void* const* d_in, const int* in_sizes, int n_in,
                              void* d_out, int out_size, void* d_ws, size_t ws_size,
                              hipStream_t stream) {
  const int*           tok  = (const int*)d_in[0];
  const unsigned char* mask = (const unsigned char*)d_in[1];   // bool8 mask
  const float* emb   = (const float*)d_in[2];
  const float* Wf    = (const float*)d_in[3];
  const float* Uf    = (const float*)d_in[4];
  const float* bfin  = (const float*)d_in[5];
  const float* bfrc  = (const float*)d_in[6];
  const float* Wb    = (const float*)d_in[7];
  const float* Ub    = (const float*)d_in[8];
  const float* bbin  = (const float*)d_in[9];
  const float* bbrc  = (const float*)d_in[10];
  (void)in_sizes; (void)n_in; (void)ws_size;

  char* ws = (char*)d_ws;
  size_t off = 0;
  auto carve = [&](size_t bytes) -> char* {
    char* p = ws + off;
    off += (bytes + 255) & ~(size_t)255;
    return p;
  };
  const size_t bpk_elems = (size_t)64 * 3 * KT_A * 512;            // 4.72M bf16 / dir
  unsigned int* ctr = (unsigned int*)carve(2 * sizeof(unsigned int));
  uint16_t* xb   = (uint16_t*)carve((size_t)BB * TT * EMBD * 2);   // 16.8 MB
  uint16_t* Bpf  = (uint16_t*)carve(bpk_elems * 2);                // 9.4 MB
  uint16_t* Bpb  = (uint16_t*)carve(bpk_elems * 2);                // 9.4 MB
  float*    hf   = (float*)   carve((size_t)2 * 2 * BB * HH * 4);  // carry, dbl-buffered
  uint16_t* hbf  = (uint16_t*)carve((size_t)2 * 2 * BB * HH * 2);

  // zero output (both dirs accumulate), h state (h0 = 0), barrier counters
  k_zero_f<<<4096, 256, 0, stream>>>((float*)d_out, (size_t)out_size);
  k_zero_f<<<64,   256, 0, stream>>>(hf, (size_t)2 * 2 * BB * HH);
  k_zero_f<<<64,   256, 0, stream>>>((float*)hbf, (size_t)2 * 2 * BB * HH / 2);
  k_zero_f<<<1,    256, 0, stream>>>((float*)ctr, 2);

  // weights -> fragment-linear packed bf16 (L2-resident, 9.4 MB per direction)
  k_pack_b<<<4096, 256, 0, stream>>>(Uf, Wf, Bpf);
  k_pack_b<<<4096, 256, 0, stream>>>(Ub, Wb, Bpb);

  // embedding gather -> bf16 activations
  k_gather_x<<<8192, 256, 0, stream>>>(tok, emb, xb);

  // persistent bidirectional GRU: 16 WGs per direction, step-synchronized
  dim3 grid(NWG, 2);
  k_bigru<<<grid, 256, 0, stream>>>(xb, Bpf, Bpb,
                                    bfin, bfrc, bbin, bbrc,
                                    mask, hf, hbf, (float*)d_out, ctr);
}
